// MessagePassing_85143431676502
// MI455X (gfx1250) — matrix operand order repaired and checked by reading, not compiled, not run on hardware
//
#include <hip/hip_runtime.h>

typedef __attribute__((ext_vector_type(16))) _Float16 v16h;
typedef __attribute__((ext_vector_type(8)))  _Float16 v8h;
typedef __attribute__((ext_vector_type(8)))  float    v8f;
typedef __attribute__((ext_vector_type(4)))  int      v4i;

#define V_NODES  50000
#define D_FEAT   128
#define H_FEAT   128
#define N_LAYERS 4
#define N_EDGES  160000
#define TILE_E   64

#if __has_builtin(__builtin_amdgcn_global_load_async_to_lds_b128)
#define HAVE_ASYNC_LDS 1
#else
#define HAVE_ASYNC_LDS 0
#endif

#define AS_GLOBAL __attribute__((address_space(1)))
#define AS_LDS    __attribute__((address_space(3)))

__device__ __forceinline__ void wait_asynccnt0() {
#if __has_builtin(__builtin_amdgcn_s_wait_asynccnt)
  __builtin_amdgcn_s_wait_asynccnt(0);
#else
  asm volatile("s_wait_asynccnt 0" ::: "memory");
#endif
}

// -------------------------------------------------------------------------
// Kernel 0: embedding table f32 -> f16 (row-major, L2 resident; halves all
// downstream gather traffic and makes the gather a raw byte copy).
// -------------------------------------------------------------------------
__global__ __launch_bounds__(256) void emb_to_f16(const float* __restrict__ emb,
                                                  _Float16* __restrict__ embH) {
  int idx = blockIdx.x * 256 + threadIdx.x;   // 800000 threads
  const float* s = emb + (size_t)idx * 8;
  v8h h;
#pragma unroll
  for (int i = 0; i < 8; ++i) h[i] = (_Float16)s[i];
  *(v8h*)(embH + (size_t)idx * 8) = h;
}

// -------------------------------------------------------------------------
// Kernel 1: W1 (L,2,256,128) f32 -> W1h (L,2,128,256) f16 (N-major, K
// contiguous); W2 (L,2,128,128) f32 -> W2h transposed likewise.
// -------------------------------------------------------------------------
__global__ __launch_bounds__(256) void prep_weights(const float* __restrict__ W1,
                                                    const float* __restrict__ W2,
                                                    _Float16* __restrict__ W1h,
                                                    _Float16* __restrict__ W2h) {
  int tid = blockIdx.x * 256 + threadIdx.x;
  const int n1 = N_LAYERS * 2 * 256 * 128;
  if (tid < n1) {
    int le  = tid / (128 * 256);
    int rem = tid % (128 * 256);
    int n   = rem / 256;
    int k   = rem % 256;
    W1h[tid] = (_Float16)W1[le * 256 * 128 + k * 128 + n];
  } else {
    int t   = tid - n1;
    int le  = t / (128 * 128);
    int rem = t % (128 * 128);
    int n   = rem / 128;
    int k   = rem % 128;
    W2h[t]  = (_Float16)W2[le * 128 * 128 + k * 128 + n];
  }
}

__global__ __launch_bounds__(256) void zero_out(float* __restrict__ out) {
  int idx = blockIdx.x * 256 + threadIdx.x;
  *(float4*)(out + (size_t)idx * 4) = make_float4(0.f, 0.f, 0.f, 0.f);
}

__global__ __launch_bounds__(256) void relu_inplace(float* __restrict__ out) {
  int idx = blockIdx.x * 256 + threadIdx.x;
  float4 v = *(float4*)(out + (size_t)idx * 4);
  v.x = fmaxf(v.x, 0.f); v.y = fmaxf(v.y, 0.f);
  v.z = fmaxf(v.z, 0.f); v.w = fmaxf(v.w, 0.f);
  *(float4*)(out + (size_t)idx * 4) = v;
}

__device__ __forceinline__ v16h cat8(v8h lo, v8h hi) {
  return __builtin_shufflevector(lo, hi, 0, 1, 2, 3, 4, 5, 6, 7,
                                         8, 9, 10, 11, 12, 13, 14, 15);
}

// -------------------------------------------------------------------------
// Main fused edge-MLP + scatter. One WG = 64 edges of one layer; 8 wave32,
// wave w owns output columns [16w,16w+16). WMMA fragment layouts used here (same k order for A and B):
//  A 16x32 f16: lane&15 = row M; per lane one contiguous 32B chunk at
//               K = (lane>>4)*16.
//  B 32x16 f16: lane&15 = col N; one contiguous 32B chunk at K=(lane>>4)*16
//               (weights pre-transposed to K-contiguous).
//  C/D 16x16 f32: lane&15 = N; vgpr g -> M = (lane>>4)*8 + g.
// -------------------------------------------------------------------------
__global__ __launch_bounds__(256) void edge_mlp_scatter(
    const _Float16* __restrict__ embH,
    const int* __restrict__ adj0, const int* __restrict__ adj1,
    const int* __restrict__ adj2, const int* __restrict__ adj3,
    const _Float16* __restrict__ W1h, const _Float16* __restrict__ W2h,
    float* __restrict__ out) {
  __shared__ alignas(16) _Float16 rawA[TILE_E * 256];  // 32 KB gathered feats
  __shared__ alignas(16) _Float16 hbuf[TILE_E * 128];  // 16 KB relu(raw@W1)
  __shared__ int sIdx[TILE_E * 2];

  const int tile  = blockIdx.x;
  const int layer = blockIdx.y;
  const int* adjL = (layer == 0) ? adj0 : (layer == 1) ? adj1
                  : (layer == 2) ? adj2 : adj3;
  const int eBase = tile * TILE_E;
  const int tid   = threadIdx.x;
  const int wave  = tid >> 5;
  const int lane  = tid & 31;

  if (tid < TILE_E * 2) sIdx[tid] = adjL[eBase * 2 + tid];
  __syncthreads();

  // Gather: thread -> (edge j = tid>>2, 64-feature segment seg = tid&3).
  // Pure byte copy of f16 rows: async global->LDS (no VGPR round trip).
  {
    int j = tid >> 2, seg = tid & 3;
    int node = sIdx[j * 2 + (seg >> 1)];               // seg 0,1: src; 2,3: dst
    const _Float16* gp = embH + (size_t)node * D_FEAT + (seg & 1) * 64;
    _Float16* lp = &rawA[j * 256 + seg * 64];
#if HAVE_ASYNC_LDS
#pragma unroll
    for (int i = 0; i < 8; ++i) {
      __builtin_amdgcn_global_load_async_to_lds_b128(
          (AS_GLOBAL v4i*)(gp + i * 8), (AS_LDS v4i*)(lp + i * 8), 0, 0);
    }
    wait_asynccnt0();
#else
#pragma unroll
    for (int i = 0; i < 8; ++i)
      *(v8h*)(lp + i * 8) = *(const v8h*)(gp + i * 8);
#endif
  }
  __syncthreads();

  const int col      = lane & 15;
  const int half     = lane >> 4;
  const int nglob    = wave * 16 + col;
  const int mrowbase = half * 8;

  for (int ep = 0; ep < 2; ++ep) {
    const _Float16* B1 = W1h + (size_t)(layer * 2 + ep) * 128 * 256;
    const _Float16* B2 = W2h + (size_t)(layer * 2 + ep) * 128 * 128;

    // ---- GEMM1: rawA(64x256) @ W1(256x128) ----
    v8f acc[4] = {{0.f}, {0.f}, {0.f}, {0.f}};
#pragma unroll
    for (int kk = 0; kk < 8; ++kk) {
      v16h bfrag = *(const v16h*)(B1 + (size_t)nglob * 256 + kk * 32 + half * 16);
      v16h af[4];
#pragma unroll
      for (int mt = 0; mt < 4; ++mt) {       // batch the 8 ds_load_b128s
        const _Float16* ap = &rawA[(mt * 16 + col) * 256 + kk * 32 + half * 16];
        af[mt] = cat8(*(const v8h*)ap, *(const v8h*)(ap + 8));
      }
#pragma unroll
      for (int mt = 0; mt < 4; ++mt)         // 4 back-to-back WMMAs
        acc[mt] = __builtin_amdgcn_wmma_f32_16x16x32_f16(
            false, af[mt], false, bfrag, (short)0, acc[mt], false, false);
    }
#pragma unroll
    for (int mt = 0; mt < 4; ++mt)
#pragma unroll
      for (int g = 0; g < 8; ++g) {
        float x = acc[mt][g];
        hbuf[(mt * 16 + mrowbase + g) * 128 + nglob] =
            (_Float16)(x > 0.f ? x : 0.f);
      }
    __syncthreads();

    // ---- GEMM2: h(64x128) @ W2(128x128) ----
    v8f macc[4] = {{0.f}, {0.f}, {0.f}, {0.f}};
#pragma unroll
    for (int kk = 0; kk < 4; ++kk) {
      v16h bfrag = *(const v16h*)(B2 + (size_t)nglob * 128 + kk * 32 + half * 16);
      v16h af[4];
#pragma unroll
      for (int mt = 0; mt < 4; ++mt) {
        const _Float16* ap = &hbuf[(mt * 16 + col) * 128 + kk * 32 + half * 16];
        af[mt] = cat8(*(const v8h*)ap, *(const v8h*)(ap + 8));
      }
#pragma unroll
      for (int mt = 0; mt < 4; ++mt)
        macc[mt] = __builtin_amdgcn_wmma_f32_16x16x32_f16(
            false, af[mt], false, bfrag, (short)0, macc[mt], false, false);
    }

    // ---- Scatter-add messages into agg[target] ----
#pragma unroll
    for (int mt = 0; mt < 4; ++mt)
#pragma unroll
      for (int g = 0; g < 8; ++g) {
        int m   = mt * 16 + mrowbase + g;
        int tgt = sIdx[m * 2 + ep];
        atomicAdd(out + (size_t)tgt * H_FEAT + nglob, macc[mt][g]);
      }
    __syncthreads();  // hbuf reused next endpoint
  }
}

extern "C" void kernel_launch(void* const* d_in, const int* in_sizes, int n_in,
                              void* d_out, int out_size, void* d_ws, size_t ws_size,
                              hipStream_t stream) {
  const float* emb  = (const float*)d_in[0];
  const int*   adj0 = (const int*)d_in[1];
  const int*   adj1 = (const int*)d_in[2];
  const int*   adj2 = (const int*)d_in[3];
  const int*   adj3 = (const int*)d_in[4];
  const float* W1   = (const float*)d_in[5];
  const float* W2   = (const float*)d_in[6];
  float*       out  = (float*)d_out;

  _Float16* embH = (_Float16*)d_ws;                            // 12.8 MB
  _Float16* W1h  = embH + (size_t)V_NODES * D_FEAT;            // 512 KB
  _Float16* W2h  = W1h + (size_t)N_LAYERS * 2 * 128 * 256;     // 256 KB

  emb_to_f16<<<(V_NODES * D_FEAT) / (256 * 8), 256, 0, stream>>>(emb, embH);
  prep_weights<<<1536, 256, 0, stream>>>(W1, W2, W1h, W2h);
  zero_out<<<(V_NODES * H_FEAT) / 1024, 256, 0, stream>>>(out);

  dim3 grid(N_EDGES / TILE_E, N_LAYERS);
  edge_mlp_scatter<<<grid, 256, 0, stream>>>(embH, adj0, adj1, adj2, adj3,
                                             W1h, W2h, out);

  relu_inplace<<<(V_NODES * H_FEAT) / 1024, 256, 0, stream>>>(out);
}
